// GraphAttentionLayer_89945205113532
// MI455X (gfx1250) — compile-verified, matrix operand order checked by reading, not run on hardware
//
#include <hip/hip_runtime.h>

#define GN   8192
#define GIN  512
#define GOUT 256

typedef __attribute__((ext_vector_type(16))) _Float16 v16h;
typedef __attribute__((ext_vector_type(8)))  _Float16 v8h;
typedef __attribute__((ext_vector_type(4)))  _Float16 v4h;
typedef __attribute__((ext_vector_type(8)))  float    v8f;

struct H2 { _Float16 lo, hi; };  // packed f16 pair -> single b32 LDS store

// Fragment K mapping per ISA 16-bit layouts:
// lanes 0-15 hold K {0..7, 16..23}, lanes 16-31 hold K {8..15, 24..31}.
__device__ __forceinline__ int frag_k(int idx, int khalf) {
  return ((idx < 8) ? idx : idx + 8) + khalf;
}

// ---------------------------------------------------------------------------
// Kernel 1: h = x @ W^T  (M=8192, K=512, N=256), f16 WMMA.
// Writes hT as f16 [GOUT][GN] (transposed) so the alpha@h GEMM gets
// contiguous K-major B fragments. Block: 256 threads = 8 waves,
// block tile 64(M) x 64(N), wave tile 16x32 (2 wmma / iter).
// ---------------------------------------------------------------------------
__global__ __launch_bounds__(256) void proj_wmma(const float* __restrict__ x,
                                                 const float* __restrict__ W,
                                                 _Float16* __restrict__ hT) {
  // As: [64 m][34]  (17-dword rows -> conflict-free, pair-contiguous K)
  // Bs: [64 n][34]  K-major per column -> fragment reads pair-contiguous
  // T : [64 n][68]  epilogue transpose tile (8B-aligned chunks), aliases As+Bs
  __shared__ _Float16 smem[64 * 68];   // 4352 halves = max(As+Bs, T)
  _Float16* As = smem;
  _Float16* Bs = smem + 64 * 34;
  _Float16* T  = smem;

  const int tid  = threadIdx.x;
  const int m0   = blockIdx.y * 64;
  const int n0   = blockIdx.x * 64;
  const int lane = tid & 31;
  const int wv   = tid >> 5;
  const int mo   = 16 * (wv >> 1);     // wave M sub-tile
  const int no2  = 32 * (wv & 1);      // wave N sub-tile (2 wmma tiles)
  const int ra   = lane & 15;
  const int khalf = (lane >> 4) << 3;

  v8f c0 = {};
  v8f c1 = {};

  for (int kk = 0; kk < GIN; kk += 32) {
    __syncthreads();
    // Stage A: 64x32 tile as 1024 f32-pairs; b64 load -> cvt_pk -> b32 store.
#pragma unroll
    for (int i = 0; i < 4; ++i) {
      int p = tid + i * 256;
      int r = p >> 4, cp = (p & 15) * 2;
      float2 f = *(const float2*)&x[(size_t)(m0 + r) * GIN + kk + cp];
      *(H2*)(As + r * 34 + cp) = H2{(_Float16)f.x, (_Float16)f.y};
    }
    // Stage B (K-major): Bs[n][k] = W[n0+n][kk+k], same pairing.
#pragma unroll
    for (int i = 0; i < 4; ++i) {
      int p = tid + i * 256;
      int n = p >> 4, cp = (p & 15) * 2;
      float2 f = *(const float2*)&W[(size_t)(n0 + n) * GIN + kk + cp];
      *(H2*)(Bs + n * 34 + cp) = H2{(_Float16)f.x, (_Float16)f.y};
    }
    __syncthreads();

    v16h af, b0, b1;
#pragma unroll
    for (int idx = 0; idx < 16; ++idx) {
      int k = frag_k(idx, khalf);
      af[idx] = As[(mo + ra) * 34 + k];
      b0[idx] = Bs[(no2 + ra) * 34 + k];
      b1[idx] = Bs[(no2 + 16 + ra) * 34 + k];
    }
    c0 = __builtin_amdgcn_wmma_f32_16x16x32_f16(false, af, false, b0, (short)0, c0, false, false);
    c1 = __builtin_amdgcn_wmma_f32_16x16x32_f16(false, af, false, b1, (short)0, c1, false, false);
  }

  // Epilogue: stage D tiles into LDS as T[n][m] (packed pairs), then
  // vectorized coalesced store to hT.
  __syncthreads();
#pragma unroll
  for (int v = 0; v < 8; v += 2) {
    int m = mo + v + ((lane >> 4) << 3);
    *(H2*)(T + (no2 + ra) * 68 + m)      = H2{(_Float16)c0[v], (_Float16)c0[v + 1]};
    *(H2*)(T + (no2 + 16 + ra) * 68 + m) = H2{(_Float16)c1[v], (_Float16)c1[v + 1]};
  }
  __syncthreads();
  // 64x64 tile = 1024 chunks of 4 halves (8B): ds_load_b64 + global_store_b64.
#pragma unroll
  for (int i = 0; i < 4; ++i) {
    int cidx = tid + i * 256;
    int n = cidx >> 4, mm = (cidx & 15) * 4;
    *(v4h*)&hT[(size_t)(n0 + n) * GN + m0 + mm] = *(const v4h*)(T + n * 68 + mm);
  }
}

// ---------------------------------------------------------------------------
// Kernel 2: s_j[j] = sum_n hT[n][j] * a[GOUT+n]   (coalesced over j per n)
// ---------------------------------------------------------------------------
__global__ __launch_bounds__(256) void sj_kernel(const _Float16* __restrict__ hT,
                                                 const float* __restrict__ a,
                                                 float* __restrict__ sj) {
  int j = blockIdx.x * blockDim.x + threadIdx.x;
  float s = 0.f;
#pragma unroll 4
  for (int n = 0; n < GOUT; ++n)
    s += (float)hT[(size_t)n * GN + j] * a[GOUT + n];
  sj[j] = s;
}

// ---------------------------------------------------------------------------
// Kernel 3: per-row online softmax stats over masked s_j.
// One 256-thread workgroup per row; single coalesced pass over adj row.
// ---------------------------------------------------------------------------
__global__ __launch_bounds__(256) void row_stats(const int* __restrict__ adj,
                                                 const float* __restrict__ sj,
                                                 float* __restrict__ rmax,
                                                 float* __restrict__ rsum) {
  __shared__ float sm[256];
  __shared__ float sz[256];
  const int i = blockIdx.x;
  const int tid = threadIdx.x;
  const size_t rowo = (size_t)i * GN;
  const float NEG_INF = -__builtin_inff();

  float m = NEG_INF, z = 0.f;
  for (int jb = tid * 4; jb < GN; jb += 1024) {   // int4/float4 per thread
    int4   av = *(const int4*)&adj[rowo + jb];
    float4 sv = *(const float4*)&sj[jb];
    const int   ai[4] = {av.x, av.y, av.z, av.w};
    const float si[4] = {sv.x, sv.y, sv.z, sv.w};
#pragma unroll
    for (int t = 0; t < 4; ++t) {
      if (ai[t] > 0) {
        float s = si[t];
        if (s > m) { z = z * __expf(m - s) + 1.f; m = s; }  // m==-inf: z==0
        else       { z += __expf(s - m); }
      }
    }
  }
  sm[tid] = m; sz[tid] = z;
  __syncthreads();
  for (int off = 128; off > 0; off >>= 1) {
    if (tid < off) {
      float m2 = sm[tid + off], z2 = sz[tid + off];
      float M = fmaxf(m, m2);
      float zz = 0.f;
      if (z  > 0.f) zz += z  * __expf(m  - M);
      if (z2 > 0.f) zz += z2 * __expf(m2 - M);
      m = M; z = zz;
      sm[tid] = m; sz[tid] = z;
    }
    __syncthreads();
  }
  if (tid == 0) { rmax[i] = m; rsum[i] = z; }
}

// ---------------------------------------------------------------------------
// Kernel 4: fused alpha + out = alpha @ h + ELU.
// Block: 512 threads = 16 waves, 16 rows x all 8192 j.
// Per iteration (128 j): each thread handles 4 consecutive j (int4 adj load,
// float4 sj load, float4 alpha store, 8B f16 LDS tile store), then each wave
// does four v_wmma_f32_16x16x32_f16 for its 16-column output tile.
// Double-buffered tile -> ONE barrier per iteration (64 total).
// hT (4 MB) stays resident in the 192 MB L2.
// ---------------------------------------------------------------------------
__global__ __launch_bounds__(512) void alpha_out_wmma(const int* __restrict__ adj,
                                                      const float* __restrict__ sj,
                                                      const float* __restrict__ rmax,
                                                      const float* __restrict__ rsum,
                                                      const _Float16* __restrict__ hT,
                                                      float* __restrict__ out_elu,
                                                      float* __restrict__ out_alpha) {
  // 16 rows x 128 j per tile; row pitch 132 halves (66 dwords: lane stride
  // 2ra mod 64 across 16 rows -> conflict-free), double buffered.
  __shared__ _Float16 ath[2][16 * 132];

  const int tid  = threadIdx.x;
  const int r    = tid >> 5;        // row-in-tile (== wave id)
  const int jj   = tid & 31;
  const int lane = tid & 31;
  const int i0   = blockIdx.x * 16;
  const int row  = i0 + r;

  const float mrow  = rmax[row];
  const float izrow = 1.f / rsum[row];

  const int n0w   = 16 * r;            // this wave's output column tile
  const int ra    = lane & 15;
  const int khalf = (lane >> 4) << 3;
  const _Float16* bbase = hT + (size_t)(n0w + ra) * GN;
  const int*   adjrow = adj + (size_t)row * GN;
  float*       arow   = out_alpha + (size_t)row * GN;

  v8f c = {};
  int ph = 0;
  for (int j0 = 0; j0 < GN; j0 += 128, ph ^= 1) {
    const int jq = j0 + jj * 4;          // this thread's 4 consecutive j
    __builtin_prefetch(adjrow + j0 + 128, 0, 1);
    int4   av = *(const int4*)&adjrow[jq];
    float4 sv = *(const float4*)&sj[jq];
    float4 al;
    al.x = (av.x > 0) ? __expf(sv.x - mrow) * izrow : 0.f;
    al.y = (av.y > 0) ? __expf(sv.y - mrow) * izrow : 0.f;
    al.z = (av.z > 0) ? __expf(sv.z - mrow) * izrow : 0.f;
    al.w = (av.w > 0) ? __expf(sv.w - mrow) * izrow : 0.f;
    *(float4*)&arow[jq] = al;            // alpha output (f32, 16B coalesced)
    v4h alh = { (_Float16)al.x, (_Float16)al.y, (_Float16)al.z, (_Float16)al.w };
    *(v4h*)&ath[ph][r * 132 + jj * 4] = alh;   // fragment source (f16, 8B)
    __syncthreads();  // single barrier: double buffering covers WAR

    const _Float16* arow_lds = &ath[ph][ra * 132];
#pragma unroll
    for (int t = 0; t < 4; ++t) {
      v16h a, b;
#pragma unroll
      for (int idx = 0; idx < 16; ++idx)
        a[idx] = arow_lds[32 * t + frag_k(idx, khalf)];
      // B fragment: contiguous along K (transposed hT) -> 16B global loads.
      v8h blo = *(const v8h*)(bbase + j0 + 32 * t + khalf);
      v8h bhi = *(const v8h*)(bbase + j0 + 32 * t + khalf + 16);
#pragma unroll
      for (int q = 0; q < 8; ++q) { b[q] = blo[q]; b[q + 8] = bhi[q]; }
      c = __builtin_amdgcn_wmma_f32_16x16x32_f16(false, a, false, b, (short)0, c, false, false);
    }
  }

  // ELU epilogue; D layout: M = vgpr + 8*(lane/16), N = lane%16.
#pragma unroll
  for (int v = 0; v < 8; ++v) {
    int m   = v + ((lane >> 4) << 3);
    int col = n0w + ra;
    float val = c[v];
    val = (val > 0.f) ? val : (__expf(val) - 1.f);
    out_elu[(size_t)(i0 + m) * GOUT + col] = val;
  }
}

// ---------------------------------------------------------------------------
extern "C" void kernel_launch(void* const* d_in, const int* in_sizes, int n_in,
                              void* d_out, int out_size, void* d_ws, size_t ws_size,
                              hipStream_t stream) {
  (void)in_sizes; (void)n_in; (void)out_size; (void)ws_size;
  const float* x   = (const float*)d_in[0];   // [8192, 512]
  const int*   adj = (const int*)d_in[1];     // [8192, 8192]
  const float* W   = (const float*)d_in[2];   // [256, 512]
  const float* a   = (const float*)d_in[3];   // [512]

  float* out       = (float*)d_out;               // elu(out): 8192*256
  float* out_alpha = out + (size_t)GN * GOUT;     // alpha:    8192*8192

  char* ws = (char*)d_ws;
  _Float16* hT = (_Float16*)ws;                   // 4 MB: f16 [256][8192]
  float* sj    = (float*)(ws + (size_t)GOUT * GN * sizeof(_Float16));
  float* rmax  = sj + GN;
  float* rsum  = rmax + GN;

  proj_wmma<<<dim3(GOUT / 64, GN / 64), 256, 0, stream>>>(x, W, hT);
  sj_kernel<<<GN / 256, 256, 0, stream>>>(hT, a, sj);
  row_stats<<<GN, 256, 0, stream>>>(adj, sj, rmax, rsum);
  alpha_out_wmma<<<GN / 16, 512, 0, stream>>>(adj, sj, rmax, rsum, hT, out, out_alpha);
}
